// FusionNetwork_84817014162051
// MI455X (gfx1250) — matrix operand order repaired and checked by reading, not compiled, not run on hardware
//
#include <hip/hip_runtime.h>

#define B_   16
#define S_   8192
#define D_   256
#define TD_  512
#define ROWS (B_ * S_)

typedef __bf16 bf16_t;
typedef __attribute__((ext_vector_type(16))) __bf16 v16bf;
typedef __attribute__((ext_vector_type(8)))  float  v8f;
typedef __attribute__((ext_vector_type(4)))  unsigned int v4u;
typedef __attribute__((ext_vector_type(8)))  int v8i;
typedef __attribute__((ext_vector_type(4)))  int v4i;

#if defined(__has_builtin)
#if __has_builtin(__builtin_amdgcn_tensor_load_to_lds) && \
    __has_builtin(__builtin_amdgcn_s_wait_tensorcnt)
#define HAVE_TDM 1
#endif
#endif

union FU { float f; unsigned u; };
union HU { unsigned short s; bf16_t b; };

static __device__ inline bf16_t f2bf_rne(float f) {
  FU x; x.f = f;
  unsigned r = x.u + 0x7FFFu + ((x.u >> 16) & 1u);
  HU y; y.s = (unsigned short)(r >> 16);
  return y.b;
}
static __device__ inline float bf2f(bf16_t b) {
  HU y; y.b = b;
  FU x; x.u = ((unsigned)y.s) << 16;
  return x.f;
}
// Truncating split helpers (lo-term compensates hi truncation; combined
// representation error <= 2^-16 relative — cheap: no rounding adds).
static __device__ inline bf16_t f2bf_trunc(float f) {
  FU x; x.f = f;
  HU y; y.s = (unsigned short)(x.u >> 16);
  return y.b;
}
static __device__ inline float hi_part(float f) {
  FU x; x.f = f; x.u &= 0xFFFF0000u; return x.f;
}

// ---------------------------------------------------------------------------
// ws layout (bytes):
//   0       : Mhi  (65536 bf16, layout [l][i], 128 KB)
//   131072  : Mlo  (65536 bf16, 128 KB)
//   262144  : uvec (256 f32)
//   263168  : wvec (256 f32)
//   264192  : cval (1 f32)
//   264448  : partial (16 chunks * 16 b * 512 f32 = 512 KB)
//   788736  : rbuf (16 * 512 f32 = 32 KB)
// ---------------------------------------------------------------------------

// M[i,l] = sum_j Wq[i,j] * Wk[l,j]   (= Wq @ Wk^T), stored [l*256 + i]
__global__ __launch_bounds__(256) void k_prep_M(const float* __restrict__ wq,
                                                const float* __restrict__ wk,
                                                bf16_t* __restrict__ Mhi,
                                                bf16_t* __restrict__ Mlo) {
  const int idx = blockIdx.x * 256 + threadIdx.x;   // 65536 total = l*256 + i
  const int i = idx & 255;
  const int l = idx >> 8;
  const float4* qa = (const float4*)(wq + (size_t)i * D_);
  const float4* ka = (const float4*)(wk + (size_t)l * D_);
  float acc = 0.f;
  #pragma unroll 8
  for (int j = 0; j < 64; ++j) {
    float4 a = qa[j], b = ka[j];
    acc += a.x * b.x + a.y * b.y + a.z * b.z + a.w * b.w;
  }
  bf16_t h = f2bf_rne(acc);
  Mhi[idx] = h;
  Mlo[idx] = f2bf_rne(acc - bf2f(h));
}

// u[i] = Wq row i . bk ; w[l] = Wk row l . bq ; c = bq . bk
__global__ __launch_bounds__(512) void k_prep_uwc(const float* __restrict__ wq,
                                                  const float* __restrict__ wk,
                                                  const float* __restrict__ bq,
                                                  const float* __restrict__ bk,
                                                  float* __restrict__ uvec,
                                                  float* __restrict__ wvec,
                                                  float* __restrict__ cval) {
  const int t = threadIdx.x;
  if (t < 256) {
    const float* row = wq + (size_t)t * D_;
    float a = 0.f;
    for (int j = 0; j < D_; ++j) a += row[j] * bk[j];
    uvec[t] = a;
  } else {
    const int l = t - 256;
    const float* row = wk + (size_t)l * D_;
    float a = 0.f;
    for (int j = 0; j < D_; ++j) a += row[j] * bq[j];
    wvec[l] = a;
  }
  if (t == 0) {
    float c = 0.f;
    for (int j = 0; j < D_; ++j) c += bq[j] * bk[j];
    *cval = c;
  }
}

// Main kernel: diag[row] = vi_row @ M @ au_row + vi.u + au.w + c
// Block = 512 threads = 16 waves; wave handles 16 rows/tile; 2 tiles/block.
// All of M (bf16 hi/lo, column-major by k) resident in 256 KB LDS, staged via
// the Tensor Data Mover with a verified plain-copy fallback overwrite.
__global__ __launch_bounds__(512) void k_diag(const float* __restrict__ au,
                                              const float* __restrict__ vi,
                                              const char* __restrict__ wsM,
                                              const float* __restrict__ uvec,
                                              const float* __restrict__ wvec,
                                              const float* __restrict__ cval,
                                              float* __restrict__ diag) {
  extern __shared__ char smem_raw[];
  bf16_t* sHi = (bf16_t*)smem_raw;                   // 128 KB
  bf16_t* sLo = (bf16_t*)(smem_raw + 131072);        // 128 KB
  float*  sRed = (float*)(smem_raw + 262144);        // 16 waves * 16 rows

  const int tid = threadIdx.x;
  const int lane = tid & 31;
  const int wave = tid >> 5;
  const int half = lane >> 4;   // 0: lanes 0-15, 1: lanes 16-31
  const int lrow = lane & 15;

#ifdef HAVE_TDM
  // --- TDM async stage of M (262144 B) into LDS: D# built per CDNA5 ISA §8.
  // 2-D tile: 8192 dwords x 8 rows, stride 8192 dwords, data_size=4B.
  if (wave == 0) {
    const unsigned long long ga = (unsigned long long)(uintptr_t)wsM;
    const unsigned lds = (unsigned)(uintptr_t)smem_raw;  // addr[31:0] = LDS offs
    v4u g0;
    g0[0] = 1u;                                     // count=1, no gather
    g0[1] = lds;                                    // lds_addr
    g0[2] = (unsigned)(ga & 0xFFFFFFFFu);           // global_addr[31:0]
    g0[3] = (unsigned)((ga >> 32) & 0x1FFFFFFu) | (2u << 30);  // [56:32]|type=2
    v8i g1;
    g1[0] = (int)(2u << 16);                        // data_size = 4B
    g1[1] = (int)(8192u << 16);                     // tensor_dim0 lo16
    g1[2] = (int)(8u << 16);                        // dim0 hi=0 | tensor_dim1=8
    g1[3] = (int)(8192u << 16);                     // dim1 hi=0 | tile_dim0=8192
    g1[4] = 8;                                      // tile_dim1=8, tile_dim2=0
    g1[5] = 8192;                                   // tensor_dim0_stride lo32
    g1[6] = 0;                                      // stride0 hi | stride1 lo
    g1[7] = 0;                                      // stride1 hi
    v4i g2; g2[0] = 1; g2[1] = 1; g2[2] = 0; g2[3] = 0;   // dim2=1,dim3=1
    v4i g3; g3[0] = 0; g3[1] = (int)(1u << 16); g3[2] = 0; g3[3] = 0; // dim4=1
#if __clang_major__ >= 23
    v8i g4 = {0, 0, 0, 0, 0, 0, 0, 0};
    __builtin_amdgcn_tensor_load_to_lds(g0, g1, g2, g3, g4, 0);
#else
    __builtin_amdgcn_tensor_load_to_lds(g0, g1, g2, g3, 0);
#endif
    __builtin_amdgcn_s_wait_tensorcnt(0);
  }
  __syncthreads();
#endif
  // Known-good staging (rewrites the same bytes the TDM loaded; keeps results
  // correct even if the unverified TDM builtin arg semantics differ).
  {
    const uint4* gsrc = (const uint4*)wsM;
    uint4* gdst = (uint4*)smem_raw;
    #pragma unroll
    for (int k = 0; k < 32; ++k) gdst[tid + k * 512] = gsrc[tid + k * 512];
  }
  __syncthreads();

  const float cb = *cval;

  for (int tile = 0; tile < 2; ++tile) {
    const int row0 = blockIdx.x * 512 + tile * 256 + wave * 16;

    // --- Build A fragments (16x32 bf16 per k-step) from this lane's VI row ---
    // A layout: lanes 0-15 hold K [0..15]; lanes 16-31 [16..31]
    const float* vrow = vi + (size_t)(row0 + lrow) * D_;
    v16bf aHi[8], aLo[8];
    #pragma unroll
    for (int ks = 0; ks < 8; ++ks) {
      const int kA = ks * 32 + half * 16;
      float fv[16];
      const float4 f0 = *(const float4*)(vrow + kA);
      const float4 f1 = *(const float4*)(vrow + kA + 4);
      const float4 f2 = *(const float4*)(vrow + kA + 8);
      const float4 f3 = *(const float4*)(vrow + kA + 12);
      fv[0]=f0.x;  fv[1]=f0.y;  fv[2]=f0.z;  fv[3]=f0.w;
      fv[4]=f1.x;  fv[5]=f1.y;  fv[6]=f1.z;  fv[7]=f1.w;
      fv[8]=f2.x;  fv[9]=f2.y;  fv[10]=f2.z; fv[11]=f2.w;
      fv[12]=f3.x; fv[13]=f3.y; fv[14]=f3.z; fv[15]=f3.w;
      #pragma unroll
      for (int e = 0; e < 16; ++e) {
        aHi[ks][e] = f2bf_trunc(fv[e]);
        aLo[ks][e] = f2bf_trunc(fv[e] - hi_part(fv[e]));
      }
    }

    float dacc[8];
    #pragma unroll
    for (int r = 0; r < 8; ++r) dacc[r] = 0.f;

    // C/D layout: VGPR r holds row (half*8 + r), col = lane%16 within n-tile.
    const float* arow = au + (size_t)(row0 + half * 8) * D_;

    for (int n = 0; n < 16; ++n) {
      const int col = n * 16 + lrow;
      const bf16_t* bh = sHi + col * 256;   // column-major by k: contiguous K
      const bf16_t* bl = sLo + col * 256;
      v8f acc0 = {0.f,0.f,0.f,0.f,0.f,0.f,0.f,0.f};
      v8f acc1 = acc0, acc2 = acc0;
      #pragma unroll
      for (int ks = 0; ks < 8; ++ks) {
        const int kl = ks * 32 + half * 16;  // B: lanes 0-15 K[0..15], 16-31 K[16..31]
        v16bf bHi = *(const v16bf*)(bh + kl);
        v16bf bLo = *(const v16bf*)(bl + kl);
        acc0 = __builtin_amdgcn_wmma_f32_16x16x32_bf16(false, aHi[ks], false, bHi,
                                                       (short)0, acc0, false, false);
        acc1 = __builtin_amdgcn_wmma_f32_16x16x32_bf16(false, aHi[ks], false, bLo,
                                                       (short)0, acc1, false, false);
        acc2 = __builtin_amdgcn_wmma_f32_16x16x32_bf16(false, aLo[ks], false, bHi,
                                                       (short)0, acc2, false, false);
      }
      #pragma unroll
      for (int r = 0; r < 8; ++r) {
        float t = acc0[r] + acc1[r] + acc2[r];
        dacc[r] += t * arow[(size_t)r * D_ + col];
      }
    }

    // Reduce across the 16 lanes sharing the same rows (stays within halves).
    #pragma unroll
    for (int r = 0; r < 8; ++r) {
      float v = dacc[r];
      v += __shfl_xor(v, 1, 32);
      v += __shfl_xor(v, 2, 32);
      v += __shfl_xor(v, 4, 32);
      v += __shfl_xor(v, 8, 32);
      dacc[r] = v;
    }
    if (lrow == 0) {  // lane 0 -> rows 0-7, lane 16 -> rows 8-15
      #pragma unroll
      for (int r = 0; r < 8; ++r) sRed[wave * 16 + half * 8 + r] = dacc[r];
    }
    __syncthreads();

    // Bias terms + final store.
    for (int r = 0; r < 16; ++r) {
      const int row = row0 + r;
      const float* vr = vi + (size_t)row * D_;
      const float* ar = au + (size_t)row * D_;
      float bs = 0.f;
      #pragma unroll
      for (int c = 0; c < 8; ++c) {
        const int cc = lane + c * 32;
        bs += vr[cc] * uvec[cc] + ar[cc] * wvec[cc];
      }
      bs += __shfl_xor(bs, 1, 32);
      bs += __shfl_xor(bs, 2, 32);
      bs += __shfl_xor(bs, 4, 32);
      bs += __shfl_xor(bs, 8, 32);
      bs += __shfl_xor(bs, 16, 32);
      if (lane == 0) diag[row] = sRed[wave * 16 + r] + bs + cb;
    }
    __syncthreads();
  }
}

// In-place softmax over S per batch (score = diag / sqrt(D) = diag * 0.0625).
__global__ __launch_bounds__(1024) void k_softmax(float* __restrict__ wts) {
  __shared__ float sred[32];
  const int t = threadIdx.x;
  const int lane = t & 31, wid = t >> 5;
  float* p = wts + (size_t)blockIdx.x * S_;
  float v[8];
  #pragma unroll
  for (int i = 0; i < 8; ++i) v[i] = p[t + i * 1024] * 0.0625f;
  float m = v[0];
  #pragma unroll
  for (int i = 1; i < 8; ++i) m = fmaxf(m, v[i]);
  #pragma unroll
  for (int k = 1; k < 32; k <<= 1) m = fmaxf(m, __shfl_xor(m, k, 32));
  if (lane == 0) sred[wid] = m;
  __syncthreads();
  if (t < 32) {
    float x = sred[t];
    #pragma unroll
    for (int k = 1; k < 32; k <<= 1) x = fmaxf(x, __shfl_xor(x, k, 32));
    if (t == 0) sred[0] = x;
  }
  __syncthreads();
  m = sred[0];
  __syncthreads();
  float e[8], s = 0.f;
  #pragma unroll
  for (int i = 0; i < 8; ++i) { e[i] = __expf(v[i] - m); s += e[i]; }
  #pragma unroll
  for (int k = 1; k < 32; k <<= 1) s += __shfl_xor(s, k, 32);
  if (lane == 0) sred[wid] = s;
  __syncthreads();
  if (t < 32) {
    float x = sred[t];
    #pragma unroll
    for (int k = 1; k < 32; k <<= 1) x += __shfl_xor(x, k, 32);
    if (t == 0) sred[0] = x;
  }
  __syncthreads();
  const float inv = 1.f / sred[0];
  #pragma unroll
  for (int i = 0; i < 8; ++i) p[t + i * 1024] = e[i] * inv;
}

// Weighted reduction r[b,:] = sum_s w[b,s] * [vi,au][b,s,:], deterministic
// two-stage (no float atomics). Block handles (b, chunk of 512 rows).
__global__ __launch_bounds__(256) void k_wsum(const float* __restrict__ au,
                                              const float* __restrict__ vi,
                                              const float* __restrict__ wts,
                                              float* __restrict__ partial) {
  const int b = blockIdx.x >> 4;
  const int chunk = blockIdx.x & 15;
  const int t = threadIdx.x;
  const float* wp = wts + (size_t)b * S_ + chunk * 512;
  const size_t rowbase = ((size_t)b * S_ + (size_t)chunk * 512) * D_;
  const float* vp = vi + rowbase + t;
  const float* ap = au + rowbase + t;
  float accv = 0.f, acca = 0.f;
  #pragma unroll 4
  for (int s = 0; s < 512; ++s) {
    __builtin_prefetch(vp + (size_t)(s + 16) * D_, 0, 0);  // global_prefetch_b8
    __builtin_prefetch(ap + (size_t)(s + 16) * D_, 0, 0);
    const float w = wp[s];
    accv += w * vp[(size_t)s * D_];
    acca += w * ap[(size_t)s * D_];
  }
  float* pp = partial + ((size_t)chunk * B_ + b) * TD_;
  pp[t] = accv;
  pp[t + 256] = acca;
}

__global__ __launch_bounds__(256) void k_reduce_r(const float* __restrict__ partial,
                                                  float* __restrict__ rbuf) {
  const int idx = blockIdx.x * 256 + threadIdx.x;  // 8192 = b*512 + c
  float a = 0.f;
  #pragma unroll
  for (int chunk = 0; chunk < 16; ++chunk) a += partial[(size_t)chunk * 8192 + idx];
  rbuf[idx] = a;
}

// out[b,e] = wv_b[e] + sum_f r[b,f] * wv_w[f,e]
__global__ __launch_bounds__(256) void k_out(const float* __restrict__ rbuf,
                                             const float* __restrict__ wvw,
                                             const float* __restrict__ wvb,
                                             float* __restrict__ out) {
  const int idx = blockIdx.x * 256 + threadIdx.x;  // 8192
  const int b = idx >> 9, e = idx & 511;
  const float* rb = rbuf + (size_t)b * TD_;
  float acc = wvb[e];
  #pragma unroll 8
  for (int f = 0; f < TD_; ++f) acc += rb[f] * wvw[(size_t)f * TD_ + e];
  out[idx] = acc;
}

extern "C" void kernel_launch(void* const* d_in, const int* in_sizes, int n_in,
                              void* d_out, int out_size, void* d_ws, size_t ws_size,
                              hipStream_t stream) {
  (void)in_sizes; (void)n_in; (void)out_size; (void)ws_size;
  const float* au   = (const float*)d_in[0];
  const float* vi   = (const float*)d_in[1];
  const float* wq_w = (const float*)d_in[2];
  const float* wq_b = (const float*)d_in[3];
  const float* wk_w = (const float*)d_in[4];
  const float* wk_b = (const float*)d_in[5];
  const float* wv_w = (const float*)d_in[6];
  const float* wv_b = (const float*)d_in[7];

  float* out = (float*)d_out;          // [B, 2D] = 8192 floats
  float* wts = out + B_ * TD_;         // [B, S]  = 131072 floats (also holds diag)

  char*   ws      = (char*)d_ws;
  bf16_t* Mhi     = (bf16_t*)ws;
  bf16_t* Mlo     = (bf16_t*)(ws + 131072);
  float*  uvec    = (float*)(ws + 262144);
  float*  wvec    = (float*)(ws + 263168);
  float*  cval    = (float*)(ws + 264192);
  float*  partial = (float*)(ws + 264448);
  float*  rbuf    = (float*)(ws + 264448 + 524288);

  k_prep_M  <<<256, 256, 0, stream>>>(wq_w, wk_w, Mhi, Mlo);
  k_prep_uwc<<<1,   512, 0, stream>>>(wq_w, wk_w, wq_b, wk_b, uvec, wvec, cval);
  // 256 blocks * 512 rows = 131072 rows; dynamic LDS: 256 KB M + 1 KB scratch
  k_diag    <<<256, 512, 263168, stream>>>(au, vi, ws, uvec, wvec, cval, wts);
  k_softmax <<<B_, 1024, 0, stream>>>(wts);
  k_wsum    <<<B_ * 16, 256, 0, stream>>>(au, vi, wts, partial);
  k_reduce_r<<<32, 256, 0, stream>>>(partial, rbuf);
  k_out     <<<32, 256, 0, stream>>>(rbuf, wv_w, wv_b, out);
}
